// AttentionBlock_40072044871789
// MI455X (gfx1250) — matrix operand order repaired and checked by reading, not compiled, not run on hardware
//
#include <hip/hip_runtime.h>

typedef __bf16 bf16;
typedef __attribute__((ext_vector_type(16))) __bf16 v16bf;
typedef __attribute__((ext_vector_type(8)))  __bf16 bf16x8;
typedef __attribute__((ext_vector_type(4)))  __bf16 bf16x4;
typedef __attribute__((ext_vector_type(8)))  float  v8f;

#define BM 128
#define BN 128
#define BK 32
#define SA (BK + 8)      // LDS row stride (elements): 80 B/row, keeps 16B chunks aligned, pads banks
#define NTHREADS 256

__device__ __forceinline__ void async_wait0() {
    asm volatile("s_wait_asynccnt 0x0" ::: "memory");
}

// --- fp32 source: global -> VGPRs (issued early, consumed after compute) ---
__device__ __forceinline__ void issue_f32(const float* __restrict__ gp, int rowStart, int kStart,
                                          int ld, int tid, float4* regs) {
#pragma unroll
    for (int it = 0; it < 4; ++it) {                 // 128x32 f32 tile, float4 per thread-slot
        int i = tid + it * NTHREADS;
        int r = i >> 3, c = (i & 7) << 2;
        regs[it] = *(const float4*)(gp + (size_t)(rowStart + r) * ld + kStart + c);
    }
}
__device__ __forceinline__ void store_bf16(int tid, const float4* regs, bf16* __restrict__ lds) {
#pragma unroll
    for (int it = 0; it < 4; ++it) {
        int i = tid + it * NTHREADS;
        int r = i >> 3, c = (i & 7) << 2;
        float4 v = regs[it];
        bf16x4 h = { (bf16)v.x, (bf16)v.y, (bf16)v.z, (bf16)v.w };
        *(bf16x4*)&lds[r * SA + c] = h;
    }
}

// --- bf16 source: GLOBAL_LOAD_ASYNC_TO_LDS_B128 (memory -> LDS, ASYNCcnt-tracked) ---
__device__ __forceinline__ void issue_async(const bf16* __restrict__ gp, int rowStart, int kStart,
                                            int ld, int tid, bf16* __restrict__ lds) {
#pragma unroll
    for (int it = 0; it < 2; ++it) {                 // 128x32 bf16 tile, 16B per thread-slot
        int i = tid + it * NTHREADS;
        int r = i >> 2, c = (i & 3) << 3;
        unsigned goff  = (unsigned)(((size_t)(rowStart + r) * ld + kStart + c) * sizeof(bf16));
        unsigned laddr = (unsigned)(uintptr_t)&lds[r * SA + c];   // low 32 bits = LDS offset
        asm volatile("global_load_async_to_lds_b128 %0, %1, %2"
                     :: "v"(laddr), "v"(goff),
                        "s"((unsigned long long)(uintptr_t)gp)
                     : "memory");
    }
}

// C[m,n] = sum_k A[m,k] * Bm[n,k]  (+ bias[n]) (* scale) (causal mask n>m -> -1e30)
// OUTM: 0 = fp32 store, 1 = bf16 store, 2 = bf16 store transposed per 1024-row batch (V^T)
template<bool AF32, bool BF32, int OUTM, bool BIAS, bool CAUSAL>
__global__ __launch_bounds__(NTHREADS)
void gemm_wmma(const void* __restrict__ A, const void* __restrict__ Bmat,
               const float* __restrict__ bias, void* __restrict__ C,
               int Kdim, int lda, int ldb, int ldc,
               long aBatch, long bBatch, long cBatch, float scale) {
    __shared__ __align__(16) bf16 sA[2][BM * SA];
    __shared__ __align__(16) bf16 sB[2][BN * SA];

    const int tid = threadIdx.x;
    const int m0 = blockIdx.y * BM;
    const int n0 = blockIdx.x * BN;
    const int z  = blockIdx.z;

    const float* Af = nullptr; const bf16* Ab = nullptr;
    const float* Bf = nullptr; const bf16* Bb = nullptr;
    if constexpr (AF32) Af = (const float*)A + (size_t)z * aBatch;
    else                Ab = (const bf16*)A  + (size_t)z * aBatch;
    if constexpr (BF32) Bf = (const float*)Bmat + (size_t)z * bBatch;
    else                Bb = (const bf16*)Bmat  + (size_t)z * bBatch;

    if constexpr (CAUSAL) {
        if (n0 >= m0 + BM) {     // fully masked tile: write mask value, skip the GEMM
            float* Cz = (float*)C + (size_t)z * cBatch;
            for (int i = tid; i < BM * BN; i += NTHREADS) {
                int r = i >> 7, c = i & (BN - 1);
                Cz[(size_t)(m0 + r) * ldc + n0 + c] = -1e30f;
            }
            return;
        }
    }

    const int lane = tid & 31;
    const int wv   = tid >> 5;
    const int wm   = wv & 3;          // 4 wave-rows  x 32
    const int wn   = wv >> 2;         // 2 wave-cols  x 64
    const int half = lane >> 4;       // K-half selector per ISA fragment layout
    const int l15  = lane & 15;

    v8f acc[2][4] = {};
    float4 ra[4], rb[4];

    // -------- prologue: stage tile 0 into buffer 0 --------
    if constexpr (AF32) issue_f32(Af, m0, 0, lda, tid, ra);
    else                issue_async(Ab, m0, 0, lda, tid, sA[0]);
    if constexpr (BF32) issue_f32(Bf, n0, 0, ldb, tid, rb);
    else                issue_async(Bb, n0, 0, ldb, tid, sB[0]);
    if constexpr (AF32) store_bf16(tid, ra, sA[0]);
    if constexpr (BF32) store_bf16(tid, rb, sB[0]);
    if constexpr (!(AF32 && BF32)) async_wait0();
    __syncthreads();

    int cur = 0;
    for (int kt = 0; kt < Kdim; kt += BK) {
        const bool hasNext = (kt + BK) < Kdim;
        if (hasNext) {            // issue next tile's fetch; it overlaps the WMMAs below
            if constexpr (AF32) issue_f32(Af, m0, kt + BK, lda, tid, ra);
            else                issue_async(Ab, m0, kt + BK, lda, tid, sA[cur ^ 1]);
            if constexpr (BF32) issue_f32(Bf, n0, kt + BK, ldb, tid, rb);
            else                issue_async(Bb, n0, kt + BK, ldb, tid, sB[cur ^ 1]);
        }

        const bf16* tA = sA[cur];
        const bf16* tB = sB[cur];
        v16bf af[2], bq[4];
#pragma unroll
        for (int mt = 0; mt < 2; ++mt) {
            int row = wm * 32 + mt * 16 + l15;
            bf16x8 lo = *(const bf16x8*)&tA[row * SA + half * 8];
            bf16x8 hi = *(const bf16x8*)&tA[row * SA + 16 + half * 8];
            af[mt] = __builtin_shufflevector(lo, hi, 0,1,2,3,4,5,6,7,8,9,10,11,12,13,14,15);
        }
#pragma unroll
        for (int nt = 0; nt < 4; ++nt) {
            int nr = wn * 64 + nt * 16 + l15;
            bf16x8 lo = *(const bf16x8*)&tB[nr * SA + half * 8];
            bf16x8 hi = *(const bf16x8*)&tB[nr * SA + 16 + half * 8];
            bq[nt] = __builtin_shufflevector(lo, hi, 0,1,2,3,4,5,6,7,8,9,10,11,12,13,14,15);
        }
#pragma unroll
        for (int mt = 0; mt < 2; ++mt)
#pragma unroll
            for (int nt = 0; nt < 4; ++nt)
                acc[mt][nt] = __builtin_amdgcn_wmma_f32_16x16x32_bf16(
                    false, af[mt], false, bq[nt], (short)0, acc[mt][nt], false, false);

        __syncthreads();          // all waves done reading buffer `cur`
        if (hasNext) {            // publish buffer cur^1
            if constexpr (AF32) store_bf16(tid, ra, sA[cur ^ 1]);
            if constexpr (BF32) store_bf16(tid, rb, sB[cur ^ 1]);
            if constexpr (!(AF32 && BF32)) async_wait0();
            __syncthreads();
            cur ^= 1;
        }
    }

    // Epilogue: C element r of v8f -> m = r + 8*half, n = l15 (ISA C/D layout)
#pragma unroll
    for (int mt = 0; mt < 2; ++mt) {
#pragma unroll
        for (int nt = 0; nt < 4; ++nt) {
            int gn = n0 + wn * 64 + nt * 16 + l15;
            float bv = 0.0f;
            if constexpr (BIAS) bv = bias[gn];
#pragma unroll
            for (int r = 0; r < 8; ++r) {
                int gm = m0 + wm * 32 + mt * 16 + half * 8 + r;
                float val = acc[mt][nt][r] * scale + bv;
                if constexpr (CAUSAL) { if (gn > gm) val = -1e30f; }
                if constexpr (OUTM == 0) {
                    ((float*)C + (size_t)z * cBatch)[(size_t)gm * ldc + gn] = val;
                } else if constexpr (OUTM == 1) {
                    ((bf16*)C + (size_t)z * cBatch)[(size_t)gm * ldc + gn] = (bf16)val;
                } else {     // transposed per-batch store: Vt[b][n][t]
                    size_t b = (size_t)(gm >> 10), t = (size_t)(gm & 1023);
                    ((bf16*)C)[(b << 20) + ((size_t)gn << 10) + t] = (bf16)val;
                }
            }
        }
    }
}

// Row softmax over 1024 fp32 scores -> bf16 probabilities. One wave per row.
__global__ __launch_bounds__(NTHREADS)
void softmax_rows(const float* __restrict__ S, bf16* __restrict__ P) {
    const int lane = threadIdx.x & 31;
    const int wv   = threadIdx.x >> 5;
    const size_t row = (size_t)blockIdx.x * 8 + wv;
    const float* s = S + (row << 10);
    float v[32];
    float mx = -3.0e38f;
#pragma unroll
    for (int j = 0; j < 32; ++j) { v[j] = s[lane + j * 32]; mx = fmaxf(mx, v[j]); }
#pragma unroll
    for (int off = 16; off > 0; off >>= 1) mx = fmaxf(mx, __shfl_xor(mx, off, 32));
    float sum = 0.0f;
#pragma unroll
    for (int j = 0; j < 32; ++j) { v[j] = __expf(v[j] - mx); sum += v[j]; }
#pragma unroll
    for (int off = 16; off > 0; off >>= 1) sum += __shfl_xor(sum, off, 32);
    const float inv = 1.0f / sum;
    bf16* p = P + (row << 10);
#pragma unroll
    for (int j = 0; j < 32; ++j) p[lane + j * 32] = (bf16)(v[j] * inv);
}

extern "C" void kernel_launch(void* const* d_in, const int* in_sizes, int n_in,
                              void* d_out, int out_size, void* d_ws, size_t ws_size,
                              hipStream_t stream) {
    (void)in_sizes; (void)n_in; (void)out_size; (void)ws_size;
    const float* x   = (const float*)d_in[0];
    const float* w_q = (const float*)d_in[1];
    const float* b_q = (const float*)d_in[2];
    const float* w_k = (const float*)d_in[3];
    const float* b_k = (const float*)d_in[4];
    const float* w_v = (const float*)d_in[5];
    const float* b_v = (const float*)d_in[6];
    const float* w_o = (const float*)d_in[7];
    const float* b_o = (const float*)d_in[8];
    float* out = (float*)d_out;

    const int Bb = 8, T = 1024, D = 1024, M = Bb * T;   // 8192
    const size_t MB = 1u << 20;
    char* ws = (char*)d_ws;
    bf16*  Q   = (bf16*)(ws);               // 16 MB  (8192 x 1024 bf16)
    bf16*  Km  = (bf16*)(ws + 16 * MB);     // 16 MB
    bf16*  Vt  = (bf16*)(ws + 32 * MB);     // 16 MB  per-batch transposed V
    float* S   = (float*)(ws + 48 * MB);    // 32 MB  (8 x 1024 x 1024 f32 scores)
    bf16*  P   = (bf16*)(ws + 80 * MB);     // 16 MB  probabilities
    bf16*  ctx = (bf16*)(ws + 96 * MB);     // 16 MB  attention output

    dim3 blk(NTHREADS);
    dim3 gProj(D / BN, M / BM, 1);          // 8 x 64
    dim3 gAttn(T / BN, T / BM, Bb);         // 8 x 8 x 8
    const long TT = (long)T * T, TD = (long)T * D;

    // 1) Q/K/V projections: x @ W^T + b  (fp32 -> bf16 staging in VGPRs, pipelined)
    gemm_wmma<true, true, 1, true, false><<<gProj, blk, 0, stream>>>(
        x, w_q, b_q, Q,  D, D, D, D, 0, 0, 0, 1.0f);
    gemm_wmma<true, true, 1, true, false><<<gProj, blk, 0, stream>>>(
        x, w_k, b_k, Km, D, D, D, D, 0, 0, 0, 1.0f);
    gemm_wmma<true, true, 2, true, false><<<gProj, blk, 0, stream>>>(
        x, w_v, b_v, Vt, D, D, D, D, 0, 0, 0, 1.0f);     // transposed epilogue -> Vt[b][d][t]

    // 2) scores = (Q @ K^T) * D^-0.5, causal, fp32   (both operands async-to-LDS)
    gemm_wmma<false, false, 0, false, true><<<gAttn, blk, 0, stream>>>(
        Q, Km, nullptr, S, D, D, D, T, TD, TD, TT, 0.03125f);

    // 3) row softmax -> bf16 P
    softmax_rows<<<dim3(M / 8), blk, 0, stream>>>(S, P);

    // 4) ctx = P @ V   (B rows are rows of Vt: contiguous; both async-to-LDS)
    gemm_wmma<false, false, 1, false, false><<<gAttn, blk, 0, stream>>>(
        P, Vt, nullptr, ctx, T, T, T, D, TT, TD, TD, 1.0f);

    // 5) out = ctx @ w_o^T + b_o, fp32  (A async-to-LDS, B via VGPR convert)
    gemm_wmma<false, true, 0, true, false><<<gProj, blk, 0, stream>>>(
        ctx, w_o, b_o, out, D, D, D, D, 0, 0, 0, 1.0f);
}
